// DETR_PartMap_12352325943704
// MI455X (gfx1250) — compile-verified, matrix operand order checked by reading, not run on hardware
//
#include <hip/hip_runtime.h>

// DETR part-map mask generation for MI455X (gfx1250, wave32).
// Output [2, B, Q, H, W] f32 = 245.8 MB of write-once data -> pure store-bandwidth
// bound (~10.5 us floor at 23.3 TB/s HBM). The mask is separable
// (row-outside OR col-outside); OR(a,b) = a + b - a*b is a rank-3 GEMM, so each
// 40x40 tile is produced by v_wmma_f32_16x16x32_f16 (K=0..2 populated, exact in
// f16/f32), staged through bank-conflict-free LDS, and streamed out with
// non-temporal global_store_b128 (write-once > L2 capacity: bypass cache rinse).

typedef __attribute__((ext_vector_type(16))) _Float16 v16h;
typedef __attribute__((ext_vector_type(8)))  float    v8f;
typedef __attribute__((ext_vector_type(4)))  float    v4f;

namespace {
constexpr int NB  = 64;    // batch
constexpr int NQ  = 300;   // queries
constexpr int HM  = 40;    // mask H
constexpr int WM  = 40;    // mask W
constexpr int PAD = 52;    // LDS row stride in floats: 8*52 % 64 == 32 -> the two
                           // half-wave row groups (M and M+8) hit disjoint banks
}

__global__ __launch_bounds__(64)
void detr_partmap_wmma(const float* __restrict__ obj,
                       const float* __restrict__ sub,
                       const long long* __restrict__ img,   // int64 (h, w) per batch
                       float* __restrict__ out)
{
  // Two wave-private mask buffers (48 padded rows each): no block barrier needed,
  // same-wave DS ordering covers the store->load dependency.
  __shared__ __align__(16) float lds[2 * 48 * PAD];

  const int bq   = blockIdx.x;         // b*NQ + q
  const int b    = bq / NQ;
  const int lane = threadIdx.x & 31;
  const int m    = threadIdx.x >> 5;   // wave 0 -> object mask, wave 1 -> human mask

  // ---- per-(b,q,mask) box bounds (uniform across the wave) ----
  const float* cp = (m == 0 ? obj : sub) + (size_t)bq * 4;
  const float cx = cp[0], cy = cp[1], bw = cp[2], bh = cp[3];
  const float sh = (float)img[2 * b + 0];   // img_h
  const float sw = (float)img[2 * b + 1];   // img_w

  // floor((coord*size)/32); /32 is an exact power-of-two scale
  int x1 = (int)floorf((cx - 0.5f * bw) * sw * (1.0f / 32.0f));
  int y1 = (int)floorf((cy - 0.5f * bh) * sh * (1.0f / 32.0f));
  int x2 = (int)floorf((cx + 0.5f * bw) * sw * (1.0f / 32.0f));
  int y2 = (int)floorf((cy + 0.5f * bh) * sh * (1.0f / 32.0f));
  x2 = min(x2, WM - 1);
  y2 = min(y2, HM - 1);

  // ---- rank-3 operands: OR(oh,ow) = oh*1 + 1*ow + oh*(-ow)  (exact in f16/f32) ----
  // f16 A 16x32 layout: lanes 0-15 hold M=lane, elements 0..2 = K=0..2 (rest zero).
  // f16 B 32x16 layout: lanes 0-15 hold N=lane, elements 0..2 = K=0..2 (rest zero).
  v16h Av[3], Bv[3];
  const bool     lo = lane < 16;
  const _Float16 h1 = (_Float16)1.0f;
  const _Float16 h0 = (_Float16)0.0f;
#pragma unroll
  for (int t = 0; t < 3; ++t) {
    const int e = t * 16 + lane;   // row index for A tile t, col index for B tile t
    _Float16 oh  = (lo && (e < y1 || e > y2)) ? h1 : h0;  // row outside box
    _Float16 ow  = (lo && (e < x1 || e > x2)) ? h1 : h0;  // col outside box
    _Float16 one = lo ? h1 : h0;
    v16h a = {};  a[0] = oh;  a[1] = one; a[2] = oh;
    v16h bb = {}; bb[0] = one; bb[1] = ow; bb[2] = (_Float16)(-ow);
    Av[t] = a; Bv[t] = bb;
  }

  // ---- 3x3 WMMA tiles (padded 48x48), scatter D into LDS row-major ----
  float* mb  = lds + m * 48 * PAD;
  const int hi8 = (lane >> 4) << 3;   // D layout: VGPR r = row r (lanes 0-15) / r+8 (16-31)
  const int nc  = lane & 15;          // D column = lane % 16
#pragma unroll
  for (int ti = 0; ti < 3; ++ti) {
#pragma unroll
    for (int tj = 0; tj < 3; ++tj) {
      v8f c = {};
      v8f d = __builtin_amdgcn_wmma_f32_16x16x32_f16(
          /*neg_a=*/false, Av[ti], /*neg_b=*/false, Bv[tj],
          /*c_mod=*/(short)0, c, /*reuse_a=*/false, /*reuse_b=*/false);
      float* tb = mb + (ti * 16 + hi8) * PAD + tj * 16 + nc;
#pragma unroll
      for (int r = 0; r < 8; ++r) tb[r * PAD] = d[r];   // fuses into ds_store_2addr_b32
    }
  }

  // ---- drain: LDS rows -> non-temporal global_store_b128 ----
  // 400 float4 per mask = 12 full wave-strided steps + a 16-lane tail.
  // Division-free indexing; unrolled so ds_load latency overlaps store issue.
  const size_t obase = ((size_t)m * NB * NQ + bq) * (size_t)(HM * WM);
  {
    int row = lane / (WM / 4);               // computed once; then stepped
    int col = (lane - row * (WM / 4)) * 4;
#pragma unroll
    for (int j = 0; j < 12; ++j) {
      v4f v = *(const v4f*)(mb + row * PAD + col);                        // ds_load_b128
      __builtin_nontemporal_store(v, (v4f*)(out + obase + row * WM + col)); // NT b128
      row += 3;                              // +32 float4 == +3 rows +2 float4
      col += 8;                              // (32 % 10) * 4
      if (col >= WM) { col -= WM; row += 1; }
    }
    if (lane < 16) {                         // i = 384..399 -> rows 38,39
      v4f v = *(const v4f*)(mb + row * PAD + col);
      __builtin_nontemporal_store(v, (v4f*)(out + obase + row * WM + col));
    }
  }
}

extern "C" void kernel_launch(void* const* d_in, const int* in_sizes, int n_in,
                              void* d_out, int out_size, void* d_ws, size_t ws_size,
                              hipStream_t stream) {
  const float*     obj = (const float*)d_in[0];      // [B,Q,4] f32
  const float*     sub = (const float*)d_in[1];      // [B,Q,4] f32
  const long long* img = (const long long*)d_in[2];  // [B,2] int64
  // d_in[3] (mask) only supplies shape in the reference; unused here.
  float* out = (float*)d_out;                        // [2,B,Q,H,W] f32

  dim3 grid(NB * NQ);   // 19200 workgroups: one (b,q) pair per block
  dim3 block(64);       // 2 waves (wave32): wave0=object mask, wave1=human mask
  hipLaunchKernelGGL(detr_partmap_wmma, grid, block, 0, stream, obj, sub, img, out);
}
